// AttentionHead_34342558499092
// MI455X (gfx1250) — matrix operand order repaired and checked by reading, not compiled, not run on hardware
//
#include <hip/hip_runtime.h>
#include <hip/hip_bf16.h>

// MI455X / gfx1250 attention head: f16 WMMA (f32 accum) flash-attention.
// Stage 1: Q/K/V projections via v_wmma_f32_16x16x32_f16 (f32 -> f16 on load).
// Stage 2: split-K (KSPLIT=2) fused scores + online softmax + PV;
//          one 16-query x 1024-key slab per wave32 -> 1024 waves.
// Stage 3: tiny combine kernel merging the two key-halves per query tile.

typedef __attribute__((ext_vector_type(16))) _Float16 v16h;
typedef __attribute__((ext_vector_type(8)))  _Float16 v8h;
typedef __attribute__((ext_vector_type(8)))  float    v8f;
typedef __attribute__((ext_vector_type(4)))  float    v4f;

#define NB   4
#define NS   2048
#define NDIN 1024
#define NDO  128
#define NEGV (-1000000000.0f)
#define KSPLIT 2
#define KSPAN (NS / KSPLIT)          // 1024 keys per wave
#define NQT  ((NB * NS) / 16)        // 512 query tiles
#define NTILE (NQT * KSPLIT)         // 1024 partial tiles

__device__ __forceinline__ v16h cat16(v8h a, v8h b) {
  return __builtin_shufflevector(a, b, 0,1,2,3,4,5,6,7,8,9,10,11,12,13,14,15);
}

// ---------------------------------------------------------------------------
// Projection: Y = X @ W^T + b, output f16.  grid=(S*B/16, 3), block=256.
// blockIdx.y selects matrix (0=Q,1=K,2=V); wave w handles col tile n0=w*16.
// Q,K stored row-major [B*S, DOUT]; V stored transposed [B, DOUT, S].
// ---------------------------------------------------------------------------
__global__ __launch_bounds__(256) void proj_kernel(
    const float* __restrict__ X,
    const float* __restrict__ Wq, const float* __restrict__ Bq,
    const float* __restrict__ Wk, const float* __restrict__ Bk,
    const float* __restrict__ Wv, const float* __restrict__ Bv,
    _Float16* __restrict__ Qh, _Float16* __restrict__ Kh,
    _Float16* __restrict__ Vt)
{
  const int lane = threadIdx.x & 31;
  const int wave = threadIdx.x >> 5;
  const int row0 = blockIdx.x * 16;          // token tile
  const int mat  = blockIdx.y;               // 0=Q 1=K 2=V
  const int n0   = wave * 16;                // output-col tile (8 waves * 16 = 128)
  const int hi   = lane >> 4;
  const int l15  = lane & 15;

  const float* Wp = (mat == 0) ? Wq : (mat == 1) ? Wk : Wv;
  const float* Bp = (mat == 0) ? Bq : (mat == 1) ? Bk : Bv;

  v8f acc = {};
  for (int c = 0; c < NDIN / 32; ++c) {
    // A fragment: X rows, interleaved A mapping {k0..k0+7, k0+16..k0+23}
    const int   ak0 = c * 32 + hi * 8;
    const float* ap = X + (size_t)(row0 + l15) * NDIN + ak0;
    v4f a0 = *(const v4f*)(ap + 0);
    v4f a1 = *(const v4f*)(ap + 4);
    v4f a2 = *(const v4f*)(ap + 16);
    v4f a3 = *(const v4f*)(ap + 20);
    v16h af;
#pragma unroll
    for (int i = 0; i < 4; ++i) {
      af[i]      = (_Float16)a0[i];
      af[4 + i]  = (_Float16)a1[i];
      af[8 + i]  = (_Float16)a2[i];
      af[12 + i] = (_Float16)a3[i];
    }
    // B fragment: B[k][n] = W[n][k]; interleaved {k0..k0+7, k0+16..k0+23}
    const int   bk0 = c * 32 + hi * 8;
    const float* bp = Wp + (size_t)(n0 + l15) * NDIN + bk0;
    v4f b0 = *(const v4f*)(bp + 0);
    v4f b1 = *(const v4f*)(bp + 4);
    v4f b2 = *(const v4f*)(bp + 16);
    v4f b3 = *(const v4f*)(bp + 20);
    v16h bf;
#pragma unroll
    for (int i = 0; i < 4; ++i) {
      bf[i]      = (_Float16)b0[i];
      bf[4 + i]  = (_Float16)b1[i];
      bf[8 + i]  = (_Float16)b2[i];
      bf[12 + i] = (_Float16)b3[i];
    }
    acc = __builtin_amdgcn_wmma_f32_16x16x32_f16(false, af, false, bf,
                                                 (short)0, acc, false, false);
  }

  const float bias = Bp[n0 + l15];
  if (mat < 2) {
    _Float16* O = (mat == 0) ? Qh : Kh;
#pragma unroll
    for (int r = 0; r < 8; ++r) {
      const int m = r + hi * 8;
      O[(size_t)(row0 + m) * NDO + n0 + l15] = (_Float16)(acc[r] + bias);
    }
  } else {
#pragma unroll
    for (int r = 0; r < 8; ++r) {
      const int t = row0 + r + hi * 8;
      const int b = t >> 11, s = t & (NS - 1);
      Vt[((size_t)b * NDO + n0 + l15) * NS + s] = (_Float16)(acc[r] + bias);
    }
  }
}

// ---------------------------------------------------------------------------
// Split-K flash attention: grid=128, block=256 (8 waves).
// Wave handles 16 queries x 1024 keys (32 chunks of 32), writes unnormalized
// partial context + per-row (max, sum) to workspace.
// Tile id t: b = t/512?  mapping below: b = t>>8, rem = t&255,
//            q0 = (rem>>1)*16, half = rem&1.
// ---------------------------------------------------------------------------
__global__ __launch_bounds__(256) void attn_kernel(
    const _Float16* __restrict__ Qh, const _Float16* __restrict__ Kh,
    const _Float16* __restrict__ Vt, const unsigned char* __restrict__ mask,
    float* __restrict__ accP, float* __restrict__ mP, float* __restrict__ lP)
{
  __shared__ float Pb[8][16 * 32];   // per-wave score tile (rows=q, cols=key)
  __shared__ float nmA[8][16];       // per-row running max (broadcast)
  __shared__ float alA[8][16];       // per-row rescale factor

  const int lane = threadIdx.x & 31;
  const int wave = threadIdx.x >> 5;
  const int tile = blockIdx.x * 8 + wave;      // 1024 partial tiles
  const int b    = tile >> 8;
  const int rem  = tile & 255;
  const int q0   = (rem >> 1) * 16;
  const int ks   = (rem & 1) * KSPAN;
  const int hi   = lane >> 4;
  const int l15  = lane & 15;
  const float scl = 0.022097086912079608f;     // 1/sqrt(S) (ref scales by sqrt(seq))

  // Q A-fragments for the whole 128-dim head, kept in VGPRs.
  v16h qa[4];
#pragma unroll
  for (int c = 0; c < 4; ++c) {
    const _Float16* qp =
        Qh + (size_t)(b * NS + q0 + l15) * NDO + c * 32 + hi * 8;
    v8h lo = *(const v8h*)qp;
    v8h hv = *(const v8h*)(qp + 16);
    qa[c] = cat16(lo, hv);
  }

  v8f acc[8];
#pragma unroll
  for (int t = 0; t < 8; ++t) acc[t] = (v8f){};
  float rowmax = -__builtin_inff();
  float rowsum = 0.0f;

  for (int k0 = ks; k0 < ks + KSPAN; k0 += 32) {
    // ---- scores: two 16-key subtiles, K=128 accumulated over 4 WMMAs ----
#pragma unroll
    for (int kt = 0; kt < 2; ++kt) {
      v8f s = {};
#pragma unroll
      for (int c = 0; c < 4; ++c) {
        const _Float16* kp =
            Kh + (size_t)(b * NS + k0 + kt * 16 + l15) * NDO + c * 32 + hi * 8;
        v8h lo = *(const v8h*)kp;
        v8h hv = *(const v8h*)(kp + 16);
        v16h kb = cat16(lo, hv);
        s = __builtin_amdgcn_wmma_f32_16x16x32_f16(false, qa[c], false, kb,
                                                   (short)0, s, false, false);
      }
      const int key = k0 + kt * 16 + l15;
      const bool mk = mask[b * NS + key] != 0;
#pragma unroll
      for (int r = 0; r < 8; ++r) {
        const float v = mk ? NEGV : s[r] * scl;
        Pb[wave][(r + hi * 8) * 32 + kt * 16 + l15] = v;
      }
    }
    __syncthreads();

    // ---- per-row max / rescale factor (row m owned by lane m) ----
    float alpha = 0.0f;
    if (lane < 16) {
      float cm = -__builtin_inff();
#pragma unroll 8
      for (int j = 0; j < 32; ++j) cm = fmaxf(cm, Pb[wave][lane * 32 + j]);
      const float nm = fmaxf(rowmax, cm);
      alpha = __expf(rowmax - nm);
      nmA[wave][lane] = nm;
      alA[wave][lane] = alpha;
      rowmax = nm;
    }
    __syncthreads();

    // ---- build P A-fragment (exp) + row partial sums ----
    const float nmv = nmA[wave][l15];
    v16h pa;
    float psum = 0.0f;
#pragma unroll
    for (int i = 0; i < 16; ++i) {
      const int j = ((i < 8) ? i : i + 8) + hi * 8;   // A-layout key mapping
      const float p = __expf(Pb[wave][l15 * 32 + j] - nmv);
      pa[i] = (_Float16)p;
      psum += p;
    }
    const float other = __shfl_xor(psum, 16, 32);
    if (lane < 16) rowsum = rowsum * alpha + psum + other;

    // ---- rescale accumulators by alpha[row] (row = vgpr index) ----
    float al[8];
#pragma unroll
    for (int r = 0; r < 8; ++r) al[r] = alA[wave][r + hi * 8];
#pragma unroll
    for (int t = 0; t < 8; ++t)
#pragma unroll
      for (int r = 0; r < 8; ++r) acc[t][r] *= al[r];

    // ---- PV: acc += P[16x32] @ V[32x128] (V^T rows are contiguous) ----
#pragma unroll
    for (int t = 0; t < 8; ++t) {
      const _Float16* vp =
          Vt + ((size_t)b * NDO + t * 16 + l15) * NS + k0 + hi * 8;
      v8h lo = *(const v8h*)vp;
      v8h hv = *(const v8h*)(vp + 16);
      v16h vb = cat16(lo, hv);
      acc[t] = __builtin_amdgcn_wmma_f32_16x16x32_f16(false, pa, false, vb,
                                                      (short)0, acc[t],
                                                      false, false);
    }
  }

  // ---- write unnormalized partials: acc [16][128], m[16], l[16] ----
  float* ap = accP + (size_t)tile * (16 * NDO);
#pragma unroll
  for (int t = 0; t < 8; ++t)
#pragma unroll
    for (int r = 0; r < 8; ++r) {
      const int m = r + hi * 8;
      ap[m * NDO + t * 16 + l15] = acc[t][r];
    }
  if (lane < 16) {
    mP[tile * 16 + lane] = rowmax;
    lP[tile * 16 + lane] = rowsum;
  }
}

// ---------------------------------------------------------------------------
// Combine the KSPLIT=2 partial tiles: out = (e1*a1 + e2*a2) / (e1*l1 + e2*l2).
// grid = 512 query tiles, block = 256; 8 elements per thread.
// ---------------------------------------------------------------------------
__global__ __launch_bounds__(256) void combine_kernel(
    const float* __restrict__ accP, const float* __restrict__ mP,
    const float* __restrict__ lP, float* __restrict__ out)
{
  const int qt = blockIdx.x;                 // 0..511
  const int b  = qt >> 7;
  const int q0 = (qt & 127) * 16;
  const int t0 = b * 256 + (qt & 127) * 2;   // key-half 0 tile
  const int t1 = t0 + 1;                     // key-half 1 tile

  for (int e = threadIdx.x; e < 16 * NDO; e += 256) {
    const int m = e >> 7;
    const int n = e & (NDO - 1);
    const float m1 = mP[t0 * 16 + m];
    const float m2 = mP[t1 * 16 + m];
    const float mm = fmaxf(m1, m2);
    const float e1 = __expf(m1 - mm);
    const float e2 = __expf(m2 - mm);
    const float l  = e1 * lP[t0 * 16 + m] + e2 * lP[t1 * 16 + m];
    const float v  = e1 * accP[(size_t)t0 * (16 * NDO) + e] +
                     e2 * accP[(size_t)t1 * (16 * NDO) + e];
    out[(size_t)(b * NS + q0 + m) * NDO + n] = v / l;
  }
}

// ---------------------------------------------------------------------------
extern "C" void kernel_launch(void* const* d_in, const int* in_sizes, int n_in,
                              void* d_out, int out_size, void* d_ws,
                              size_t ws_size, hipStream_t stream) {
  const float*         X    = (const float*)d_in[0];
  const unsigned char* mask = (const unsigned char*)d_in[1];  // jnp bool = 1B
  const float* Wq = (const float*)d_in[2];
  const float* Bq = (const float*)d_in[3];
  const float* Wk = (const float*)d_in[4];
  const float* Bk = (const float*)d_in[5];
  const float* Wv = (const float*)d_in[6];
  const float* Bv = (const float*)d_in[7];
  float* out = (float*)d_out;

  // Workspace: Q,K f16 [B*S,128] + V^T f16 [B,128,S]   (3 * 2 MB)
  //          + split-K partials: acc f32 [1024][16][128] (8 MB), m/l (128 KB)
  _Float16* Qh = (_Float16*)d_ws;
  _Float16* Kh = Qh + (size_t)NB * NS * NDO;
  _Float16* Vt = Kh + (size_t)NB * NS * NDO;
  float* accP  = (float*)(Vt + (size_t)NB * NS * NDO);
  float* mP    = accP + (size_t)NTILE * 16 * NDO;
  float* lP    = mP + (size_t)NTILE * 16;

  proj_kernel<<<dim3((NB * NS) / 16, 3), 256, 0, stream>>>(
      X, Wq, Bq, Wk, Bk, Wv, Bv, Qh, Kh, Vt);
  attn_kernel<<<dim3(NTILE / 8), 256, 0, stream>>>(
      Qh, Kh, Vt, mask, accP, mP, lP);
  combine_kernel<<<dim3(NQT), 256, 0, stream>>>(accP, mP, lP, out);
}
